// PointNet2Cls_7301444403789
// MI455X (gfx1250) — matrix operand order repaired and checked by reading, not compiled, not run on hardware
//
#include <hip/hip_runtime.h>
#include <hip/hip_bf16.h>

typedef __bf16 bf16;
typedef __attribute__((ext_vector_type(16))) __bf16 v16bf;
typedef __attribute__((ext_vector_type(8)))  __bf16 v8bf;
typedef __attribute__((ext_vector_type(8)))  float  v8f;

__device__ __forceinline__ bf16 to_bf(float f) { return (bf16)f; }

// A fragment (16x32 bf16, MxK) from row-major buffer (ld multiple of 8 bf16):
// lane half 0 holds K=0..7 (v0..3) and 16..23 (v4..7); half 1 holds 8..15, 24..31.
// Two contiguous 16B loads per lane.
__device__ __forceinline__ v16bf load_a_frag(const bf16* base, int ld, int mbase, int k0, int lane) {
  int hh = lane >> 4;
  int m  = mbase + (lane & 15);
  const bf16* row = base + (size_t)m * ld + k0 + hh * 8;
  v8bf lo = *(const v8bf*)(row);
  v8bf hi = *(const v8bf*)(row + 16);
  return __builtin_shufflevector(lo, hi, 0, 1, 2, 3, 4, 5, 6, 7,
                                 8, 9, 10, 11, 12, 13, 14, 15);
}

// B fragment from pre-swizzled weights: 32 contiguous bytes per lane.
// Layout: Wsw[(((nt*Kt + kt)*32 + lane)*16 + e)] with k = kt*32 + (e<8 ? (lane>>4)*8 + e : 16 + (lane>>4)*8 + e-8).
__device__ __forceinline__ v16bf load_b_swz(const bf16* Wsw, int Kt, int kt, int nt, int lane) {
  return *(const v16bf*)(Wsw + (((size_t)(nt * Kt + kt) * 32 + lane) << 4));
}

__device__ __forceinline__ v8f wmma_bf16(v16bf a, v16bf b, v8f c) {
  return __builtin_amdgcn_wmma_f32_16x16x32_bf16(false, a, false, b, (short)0, c, false, false);
}

// ---------------- elementwise helpers ----------------

__global__ void convert_bf16_kernel(const float* __restrict__ src, bf16* __restrict__ dst, int n) {
  int e = blockIdx.x * 256 + threadIdx.x;
  if (e < n) dst[e] = to_bf(src[e]);
}

// f32 [K,N] -> bf16 swizzled into B-fragment order, K zero-padded to Kp (mult of 32).
__global__ void convert_swz_kernel(const float* __restrict__ src, bf16* __restrict__ dst,
                                   int K, int N, int Kp) {
  int o = blockIdx.x * 256 + threadIdx.x;
  if (o >= Kp * N) return;
  int Kt = Kp >> 5;
  int frag = o >> 9, within = o & 511;
  int lane = within >> 4, e = within & 15;
  int nt = frag / Kt, kt = frag - nt * Kt;
  int hh = lane >> 4;
  int n = nt * 16 + (lane & 15);
  int k = kt * 32 + (e < 8 ? hh * 8 + e : 16 + hh * 8 + (e - 8));
  dst[o] = (k < K) ? to_bf(src[(size_t)k * N + n]) : to_bf(0.0f);
}

// ---------------- farthest point sampling ----------------
__global__ void __launch_bounds__(256) fps_kernel(const float* __restrict__ pos,
                                                  int N, int S, int* __restrict__ idx) {
  __shared__ float sDist[2048];
  __shared__ float sRD[256];
  __shared__ int   sRI[256];
  __shared__ float sC[3];
  int b = blockIdx.x, tid = threadIdx.x;
  const float* P = pos + (size_t)b * N * 3;
  for (int i = tid; i < N; i += 256) sDist[i] = 3.0e38f;
  int cur = 0;
  if (tid == 0) idx[(size_t)b * S] = 0;
  __syncthreads();
  for (int t = 1; t < S; ++t) {
    if (tid < 3) sC[tid] = P[cur * 3 + tid];
    __syncthreads();
    float cx = sC[0], cy = sC[1], cz = sC[2];
    float best = -1.0f; int bi = 0;
    for (int i = tid; i < N; i += 256) {
      float dx = P[i * 3 + 0] - cx, dy = P[i * 3 + 1] - cy, dz = P[i * 3 + 2] - cz;
      float d  = dx * dx + dy * dy + dz * dz;
      float nd = fminf(sDist[i], d);
      sDist[i] = nd;
      if (nd > best) { best = nd; bi = i; }
    }
    sRD[tid] = best; sRI[tid] = bi;
    __syncthreads();
    for (int o = 128; o > 0; o >>= 1) {
      if (tid < o) {
        float od = sRD[tid + o]; int oi = sRI[tid + o];
        if (od > sRD[tid] || (od == sRD[tid] && oi < sRI[tid])) { sRD[tid] = od; sRI[tid] = oi; }
      }
      __syncthreads();
    }
    cur = sRI[0];
    if (tid == 0) idx[(size_t)b * S + t] = cur;
    __syncthreads();
  }
}

__global__ void gather_centers_kernel(const float* __restrict__ pos, const int* __restrict__ idx,
                                      float* __restrict__ outp, int N, int S, int total) {
  int e = blockIdx.x * 256 + threadIdx.x;
  if (e >= total) return;
  int d = e % 3; int rem = e / 3; int s = rem % S; int b = rem / S;
  outp[e] = pos[((size_t)b * N + idx[(size_t)b * S + s]) * 3 + d];
}

// ---------------- fused set-abstraction module (templated dims) ----------------
// One block (4 waves) per ball: radius query (first <=64 in index order),
// chunked gather of [64,KP1] bf16 features, fully-unrolled 3-layer WMMA MLP in LDS,
// masked max-pool over neighbors.
template <int C, int KP1, int C1, int C2, int C3>
__global__ void __launch_bounds__(128) fused_sa_kernel(
    const bf16* __restrict__ x, const float* __restrict__ pos, const float* __restrict__ ctrs,
    int N, int S,
    const bf16* __restrict__ W1, const float* __restrict__ B1,
    const bf16* __restrict__ W2, const float* __restrict__ B2,
    const bf16* __restrict__ W3, const float* __restrict__ B3,
    float r2, bf16* __restrict__ xout)
{
  constexpr int LD0 = KP1 + 8;   // padded strides: +8 bf16 = 4-bank shift, keeps 16B align
  constexpr int LD1 = C1 + 8;
  constexpr int LD2 = C2 + 8;
  __shared__ bf16  sFeat[64 * LD0];
  __shared__ bf16  sH1[64 * LD1];
  __shared__ bf16  sH2[64 * LD2];
  __shared__ float sPool[4 * C3];
  __shared__ int   sNb[64];
  __shared__ int   sValid[64];
  __shared__ float sCtr[3];

  int s = blockIdx.x, b = blockIdx.y;
  int tid = threadIdx.x, lane = tid & 31, w = tid >> 5;

  if (tid < 3)  sCtr[tid] = ctrs[((size_t)b * S + s) * 3 + tid];
  if (tid < 64) { sNb[tid] = 0; sValid[tid] = 0; }
  __builtin_prefetch(W1, 0, 0);
  __builtin_prefetch(W2, 0, 0);
  __builtin_prefetch(W3, 0, 0);
  __syncthreads();

  // Ball query by wave 0: first <=64 in-radius points in index order (wave32 ballot).
  if (w == 0) {
    float cx = sCtr[0], cy = sCtr[1], cz = sCtr[2];
    int cnt = 0;
    for (int base = 0; base < N && cnt < 64; base += 32) {
      int i = base + lane;
      const float* pp = pos + ((size_t)b * N + i) * 3;
      float dx = pp[0] - cx, dy = pp[1] - cy, dz = pp[2] - cz;
      bool inr = (dx * dx + dy * dy + dz * dz) <= r2;
      unsigned mask = (unsigned)__ballot(inr);
      if (inr) {
        int slot = cnt + __popc(mask & ((1u << lane) - 1u));
        if (slot < 64) { sNb[slot] = i; sValid[slot] = 1; }
      }
      cnt += __popc(mask);
    }
  }
  __syncthreads();

  // Chunked gather: [xj(C) | pj-ctr(3) | zero pad] -> [64, KP1] bf16 (stride LD0).
  // 8-element (16B) chunks; validity hoisted per chunk; full-x chunks are vector copies.
  constexpr int Cin = C + 3;
  constexpr int NCH = KP1 / 8;
  for (int t = tid; t < 64 * NCH; t += 128) {
    int j = t / NCH, q = t - j * NCH;
    int c0 = q * 8;
    v8bf v;
#pragma unroll
    for (int e = 0; e < 8; ++e) v[e] = to_bf(0.0f);
    if (sValid[j]) {
      int pi = sNb[j];
      const bf16* xrow = x + ((size_t)b * N + pi) * C;
      if constexpr ((C % 8) == 0) {
        if (c0 + 8 <= C) {
          v = *(const v8bf*)(xrow + c0);           // aligned 16B vector copy
        } else if (c0 < Cin) {
#pragma unroll
          for (int e = 0; e < 8; ++e) {
            int c = c0 + e;
            if (c >= C && c < Cin)
              v[e] = to_bf(pos[((size_t)b * N + pi) * 3 + (c - C)] - sCtr[c - C]);
          }
        }
      } else {
        if (c0 < Cin) {
#pragma unroll
          for (int e = 0; e < 8; ++e) {
            int c = c0 + e;
            if (c < C)        v[e] = xrow[c];
            else if (c < Cin) v[e] = to_bf(pos[((size_t)b * N + pi) * 3 + (c - C)] - sCtr[c - C]);
          }
        }
      }
    }
    *(v8bf*)(sFeat + j * LD0 + c0) = v;
  }
  __syncthreads();

  int mbase = 16 * w;
  int hh = lane >> 4, col = lane & 15;
  v8f vzero = {0.f, 0.f, 0.f, 0.f, 0.f, 0.f, 0.f, 0.f};

  // Layer 1
#pragma unroll
  for (int nt = 0; nt < C1 / 16; ++nt) {
    v8f acc = vzero;
#pragma unroll
    for (int ks = 0; ks < KP1 / 32; ++ks) {
      v16bf a  = load_a_frag(sFeat, LD0, mbase, ks * 32, lane);
      v16bf bb = load_b_swz(W1, KP1 / 32, ks, nt, lane);
      acc = wmma_bf16(a, bb, acc);
    }
    float bv = B1[nt * 16 + col];
#pragma unroll
    for (int r = 0; r < 8; ++r) {
      float v = acc[r] + bv; v = v > 0.f ? v : 0.f;
      sH1[(mbase + r + (hh ? 8 : 0)) * LD1 + nt * 16 + col] = to_bf(v);
    }
  }
  __syncthreads();

  // Layer 2
#pragma unroll
  for (int nt = 0; nt < C2 / 16; ++nt) {
    v8f acc = vzero;
#pragma unroll
    for (int ks = 0; ks < C1 / 32; ++ks) {
      v16bf a  = load_a_frag(sH1, LD1, mbase, ks * 32, lane);
      v16bf bb = load_b_swz(W2, C1 / 32, ks, nt, lane);
      acc = wmma_bf16(a, bb, acc);
    }
    float bv = B2[nt * 16 + col];
#pragma unroll
    for (int r = 0; r < 8; ++r) {
      float v = acc[r] + bv; v = v > 0.f ? v : 0.f;
      sH2[(mbase + r + (hh ? 8 : 0)) * LD2 + nt * 16 + col] = to_bf(v);
    }
  }
  __syncthreads();

  // Layer 3 + masked max-pool over 64 neighbors (ReLU>=0 so invalid->0 is exact)
#pragma unroll
  for (int nt = 0; nt < C3 / 16; ++nt) {
    v8f acc = vzero;
#pragma unroll
    for (int ks = 0; ks < C2 / 32; ++ks) {
      v16bf a  = load_a_frag(sH2, LD2, mbase, ks * 32, lane);
      v16bf bb = load_b_swz(W3, C2 / 32, ks, nt, lane);
      acc = wmma_bf16(a, bb, acc);
    }
    float bv = B3[nt * 16 + col];
    float vmax = 0.0f;
#pragma unroll
    for (int r = 0; r < 8; ++r) {
      int rowg = mbase + r + (hh ? 8 : 0);
      float v = acc[r] + bv; v = v > 0.f ? v : 0.f;
      if (!sValid[rowg]) v = 0.f;
      vmax = fmaxf(vmax, v);
    }
    vmax = fmaxf(vmax, __shfl_xor(vmax, 16));
    if (lane < 16) sPool[w * C3 + nt * 16 + col] = vmax;
  }
  __syncthreads();
  for (int c = tid; c < C3; c += 128) {
    float m = fmaxf(fmaxf(sPool[c], sPool[C3 + c]),
                    fmaxf(sPool[2 * C3 + c], sPool[3 * C3 + c]));
    xout[((size_t)b * S + s) * C3 + c] = to_bf(m);
  }
}

// ---------------- tiled WMMA GEMM (+bias, +ReLU) ----------------
// Block = 8 waves computes 128x64 of out = act(A[M,K] x Wsw + bias).
// A staged in LDS (padded stride); B fragments straight from swizzled global weights.
__global__ void __launch_bounds__(256) gemm_bias_relu_kernel(
    const bf16* __restrict__ A, const bf16* __restrict__ Wsw, const float* __restrict__ bias,
    bf16* __restrict__ out, int M, int K, int N, int do_relu)
{
  __shared__ bf16 sA[128 * 40];
  int tid = threadIdx.x, lane = tid & 31, w = tid >> 5;
  int mblk = blockIdx.x * 128, nblk = blockIdx.y * 64;
  int Kt = K >> 5;
  int nt0 = nblk >> 4;
  v8f vzero = {0.f, 0.f, 0.f, 0.f, 0.f, 0.f, 0.f, 0.f};
  v8f acc[4];
#pragma unroll
  for (int i = 0; i < 4; ++i) acc[i] = vzero;

  for (int kb = 0; kb < K; kb += 32) {
#pragma unroll
    for (int i = 0; i < 16; ++i) {
      int e = tid + i * 256, r = e >> 5, k = e & 31;
      int gr = mblk + r, gk = kb + k;
      sA[r * 40 + k] = (gr < M && gk < K) ? A[(size_t)gr * K + gk] : to_bf(0.0f);
    }
    __syncthreads();
    v16bf a = load_a_frag(sA, 40, 16 * w, 0, lane);
    int kt = kb >> 5;
#pragma unroll
    for (int nt = 0; nt < 4; ++nt) {
      v16bf bb = load_b_swz(Wsw, Kt, kt, nt0 + nt, lane);
      acc[nt] = wmma_bf16(a, bb, acc[nt]);
    }
    __syncthreads();
  }
  int hh = lane >> 4, col = lane & 15;
#pragma unroll
  for (int nt = 0; nt < 4; ++nt) {
    int gc = nblk + nt * 16 + col;
    float bv = (gc < N) ? bias[gc] : 0.0f;
#pragma unroll
    for (int r = 0; r < 8; ++r) {
      int gr = mblk + 16 * w + r + (hh ? 8 : 0);
      if (gr < M && gc < N) {
        float v = acc[nt][r] + bv;
        if (do_relu) v = fmaxf(v, 0.0f);
        out[(size_t)gr * N + gc] = to_bf(v);
      }
    }
  }
}

// Build SA3 input [4096, 288] bf16 = [x2(256) | p2(3) | pad]
__global__ void build_in3_kernel(const bf16* __restrict__ x2, const float* __restrict__ p2,
                                 bf16* __restrict__ out) {
  int e = blockIdx.x * 256 + threadIdx.x;
  if (e >= 4096 * 288) return;
  int m = e / 288, c = e - m * 288;
  bf16 v = to_bf(0.0f);
  if (c < 256)      v = x2[(size_t)m * 256 + c];
  else if (c < 259) v = to_bf(p2[(size_t)m * 3 + (c - 256)]);
  out[e] = v;
}

// Global max pool: g[16,1024] = max over 256 rows of h[4096,1024]
__global__ void gpool_kernel(const bf16* __restrict__ h, float* __restrict__ g) {
  int e = blockIdx.x * 256 + threadIdx.x;
  if (e >= 16 * 1024) return;
  int b = e >> 10, c = e & 1023;
  const bf16* base = h + (size_t)b * 256 * 1024 + c;
  float m = -3.0e38f;
  for (int s = 0; s < 256; ++s) m = fmaxf(m, (float)base[(size_t)s * 1024]);
  g[e] = m;
}

// Small f32 dense layer for the classifier head (negligible FLOPs)
__global__ void small_mlp_kernel(const float* __restrict__ A, const float* __restrict__ W,
                                 const float* __restrict__ bias, float* __restrict__ out,
                                 int M, int K, int N, int do_relu) {
  int e = blockIdx.x * 256 + threadIdx.x;
  if (e >= M * N) return;
  int m = e / N, n = e - m * N;
  float acc = bias[n];
  const float* a = A + (size_t)m * K;
  for (int k = 0; k < K; ++k) acc = fmaf(a[k], W[(size_t)k * N + n], acc);
  if (do_relu) acc = fmaxf(acc, 0.0f);
  out[e] = acc;
}

// ---------------- host orchestration ----------------

extern "C" void kernel_launch(void* const* d_in, const int* in_sizes, int n_in,
                              void* d_out, int out_size, void* d_ws, size_t ws_size,
                              hipStream_t stream) {
  (void)in_sizes; (void)n_in; (void)out_size; (void)ws_size;
  const float* x    = (const float*)d_in[0];
  const float* pos  = (const float*)d_in[1];
  const float* s1w1 = (const float*)d_in[2];  const float* s1b1 = (const float*)d_in[3];
  const float* s1w2 = (const float*)d_in[4];  const float* s1b2 = (const float*)d_in[5];
  const float* s1w3 = (const float*)d_in[6];  const float* s1b3 = (const float*)d_in[7];
  const float* s2w1 = (const float*)d_in[8];  const float* s2b1 = (const float*)d_in[9];
  const float* s2w2 = (const float*)d_in[10]; const float* s2b2 = (const float*)d_in[11];
  const float* s2w3 = (const float*)d_in[12]; const float* s2b3 = (const float*)d_in[13];
  const float* s3w1 = (const float*)d_in[14]; const float* s3b1 = (const float*)d_in[15];
  const float* s3w2 = (const float*)d_in[16]; const float* s3b2 = (const float*)d_in[17];
  const float* s3w3 = (const float*)d_in[18]; const float* s3b3 = (const float*)d_in[19];
  const float* fw1  = (const float*)d_in[20]; const float* fb1  = (const float*)d_in[21];
  const float* fw2  = (const float*)d_in[22]; const float* fb2  = (const float*)d_in[23];
  const float* fw3  = (const float*)d_in[24]; const float* fb3  = (const float*)d_in[25];
  const float* l1w  = (const float*)d_in[26]; const float* l1b  = (const float*)d_in[27];
  const float* l2w  = (const float*)d_in[28]; const float* l2b  = (const float*)d_in[29];
  const float* l3w  = (const float*)d_in[30]; const float* l3b  = (const float*)d_in[31];
  const float* l4w  = (const float*)d_in[32]; const float* l4b  = (const float*)d_in[33];
  const float* l5w  = (const float*)d_in[34]; const float* l5b  = (const float*)d_in[35];

  char* ws = (char*)d_ws;
  size_t off = 0;
  auto take = [&](size_t bytes) -> char* {
    off = (off + 255) & ~(size_t)255;
    char* p = ws + off;
    off += bytes;
    return p;
  };

  int*  idx1 = (int*)  take((size_t)16 * 1024 * 4);
  int*  idx2 = (int*)  take((size_t)16 * 256 * 4);
  float* p1  = (float*)take((size_t)16 * 1024 * 3 * 4);
  float* p2  = (float*)take((size_t)16 * 256 * 3 * 4);
  bf16* xbf  = (bf16*) take((size_t)16 * 2048 * 9 * 2);
  bf16* x1   = (bf16*) take((size_t)16 * 1024 * 128 * 2);
  bf16* x2   = (bf16*) take((size_t)16 * 256 * 256 * 2);
  bf16* w1a  = (bf16*) take((size_t)32 * 64 * 2);
  bf16* w2a  = (bf16*) take((size_t)64 * 64 * 2);
  bf16* w3a  = (bf16*) take((size_t)64 * 128 * 2);
  bf16* w1b  = (bf16*) take((size_t)160 * 128 * 2);
  bf16* w2b  = (bf16*) take((size_t)128 * 128 * 2);
  bf16* w3b  = (bf16*) take((size_t)128 * 256 * 2);
  bf16* w1c  = (bf16*) take((size_t)288 * 256 * 2);
  bf16* w2c  = (bf16*) take((size_t)256 * 512 * 2);
  bf16* w3c  = (bf16*) take((size_t)512 * 1024 * 2);
  bf16* in3  = (bf16*) take((size_t)4096 * 288 * 2);
  bf16* h3a  = (bf16*) take((size_t)4096 * 256 * 2);
  bf16* h3b  = (bf16*) take((size_t)4096 * 512 * 2);
  bf16* h3c  = (bf16*) take((size_t)4096 * 1024 * 2);
  float* g   = (float*)take((size_t)16 * 1024 * 4);
  float* t1  = (float*)take((size_t)16 * 512 * 4);
  float* t2  = (float*)take((size_t)16 * 256 * 4);
  float* t3  = (float*)take((size_t)16 * 256 * 4);
  float* t4  = (float*)take((size_t)4 * 512 * 4);
  float* t5  = (float*)take((size_t)4 * 512 * 4);
  float* t6  = (float*)take((size_t)4 * 256 * 4);
  float* t7  = (float*)take((size_t)4 * 128 * 4);

  {
    int n = 16 * 2048 * 9;
    convert_bf16_kernel<<<(n + 255) / 256, 256, 0, stream>>>(x, xbf, n);
  }
  auto swz = [&](const float* src, bf16* dst, int K, int N, int Kp) {
    int n = Kp * N;
    convert_swz_kernel<<<(n + 255) / 256, 256, 0, stream>>>(src, dst, K, N, Kp);
  };
  swz(s1w1, w1a, 12, 64, 32);    swz(s1w2, w2a, 64, 64, 64);    swz(s1w3, w3a, 64, 128, 64);
  swz(s2w1, w1b, 131, 128, 160); swz(s2w2, w2b, 128, 128, 128); swz(s2w3, w3b, 128, 256, 128);
  swz(s3w1, w1c, 259, 256, 288); swz(s3w2, w2c, 256, 512, 256); swz(s3w3, w3c, 512, 1024, 512);

  // SA1: FPS(2048->1024), centers, fused ball MLP (12->64->64->128, r=0.2)
  fps_kernel<<<16, 256, 0, stream>>>(pos, 2048, 1024, idx1);
  gather_centers_kernel<<<(16 * 1024 * 3 + 255) / 256, 256, 0, stream>>>(
      pos, idx1, p1, 2048, 1024, 16 * 1024 * 3);
  fused_sa_kernel<9, 32, 64, 64, 128><<<dim3(1024, 16), 128, 0, stream>>>(
      xbf, pos, p1, 2048, 1024, w1a, s1b1, w2a, s1b2, w3a, s1b3, 0.04f, x1);

  // SA2: FPS(1024->256), centers, fused ball MLP (131->128->128->256, r=0.4)
  fps_kernel<<<16, 256, 0, stream>>>(p1, 1024, 256, idx2);
  gather_centers_kernel<<<(16 * 256 * 3 + 255) / 256, 256, 0, stream>>>(
      p1, idx2, p2, 1024, 256, 16 * 256 * 3);
  fused_sa_kernel<128, 160, 128, 128, 256><<<dim3(256, 16), 128, 0, stream>>>(
      x1, p1, p2, 1024, 256, w1b, s2b1, w2b, s2b2, w3b, s2b3, 0.16f, x2);

  // SA3 global: per-point MLP 259->256->512->1024 then max pool per batch
  build_in3_kernel<<<(4096 * 288 + 255) / 256, 256, 0, stream>>>(x2, p2, in3);
  gemm_bias_relu_kernel<<<dim3(32, 4), 256, 0, stream>>>(in3, w1c, s3b1, h3a, 4096, 288, 256, 1);
  gemm_bias_relu_kernel<<<dim3(32, 8), 256, 0, stream>>>(h3a, w2c, s3b2, h3b, 4096, 256, 512, 1);
  gemm_bias_relu_kernel<<<dim3(32, 16), 256, 0, stream>>>(h3b, w3c, s3b3, h3c, 4096, 512, 1024, 1);
  gpool_kernel<<<(16 * 1024 + 255) / 256, 256, 0, stream>>>(h3c, g);

  // Classifier head (f32, tiny)
  small_mlp_kernel<<<(16 * 512 + 255) / 256, 256, 0, stream>>>(g, l1w, l1b, t1, 16, 1024, 512, 1);
  small_mlp_kernel<<<(16 * 256 + 255) / 256, 256, 0, stream>>>(t1, l2w, l2b, t2, 16, 512, 256, 1);
  small_mlp_kernel<<<(16 * 256 + 255) / 256, 256, 0, stream>>>(t2, l3w, l3b, t3, 16, 256, 256, 0);
  // reshape [16,256] -> [4,1024] is a no-op in row-major memory
  small_mlp_kernel<<<(4 * 512 + 255) / 256, 256, 0, stream>>>(t3, fw1, fb1, t4, 4, 1024, 512, 1);
  small_mlp_kernel<<<(4 * 512 + 255) / 256, 256, 0, stream>>>(t4, fw2, fb2, t5, 4, 512, 512, 1);
  small_mlp_kernel<<<(4 * 256 + 255) / 256, 256, 0, stream>>>(t5, fw3, fb3, t6, 4, 512, 256, 1);
  small_mlp_kernel<<<(4 * 128 + 255) / 256, 256, 0, stream>>>(t6, l4w, l4b, t7, 4, 256, 128, 1);
  small_mlp_kernel<<<(4 * 128 + 255) / 256, 256, 0, stream>>>(t7, l5w, l5b, (float*)d_out, 4, 128, 128, 0);
}
